// GRU_31413390803178
// MI455X (gfx1250) — matrix operand order repaired and checked by reading, not compiled, not run on hardware
//
#include <hip/hip_runtime.h>

// ---------------------------------------------------------------------------
// Types for CDNA5 WMMA (wave32): bf16 16x16x32 -> f32 accumulate
// ---------------------------------------------------------------------------
typedef __bf16 bf16_t;
typedef __attribute__((ext_vector_type(16))) __bf16 v16bf;
typedef __attribute__((ext_vector_type(8)))  __bf16 v8bf;
typedef __attribute__((ext_vector_type(8)))  float  v8f;

static constexpr int Bb = 32;      // batch
static constexpr int Tt = 1024;    // time
static constexpr int Dd = 256;     // input dim
static constexpr int Hh = 512;     // hidden dim
static constexpr int Oo = 256;     // output dim
static constexpr int G3 = 3 * Hh;  // 1536 gate columns (r,z,n)
static constexpr int NB_SCAN = 32; // blocks in recurrent scan (one per 16-col j-tile)
static constexpr int KC = 256;     // K-chunk staged in LDS for the GEMM (32 KB panel)

#define WMMA_BF16(a, b, c) \
  __builtin_amdgcn_wmma_f32_16x16x32_bf16(false, (a), false, (b), (short)0, (c), false, false)

// A-fragment (16x32 bf16): lane holds row l%16; halves i -> K = i + 16g (contiguous 32B)
__device__ __forceinline__ v16bf load_a_frag(const bf16_t* rowk, int g) {
  return *(const v16bf*)(rowk + 16 * g);
}
// B-fragment (32x16 bf16): lane holds col l%16; halves i -> K = i + 16g (contiguous 32B)
__device__ __forceinline__ v16bf load_b_frag(const bf16_t* rowk, int g) {
  return *(const v16bf*)(rowk + 16 * g);
}

__device__ __forceinline__ float fast_sigmoid(float x) {
  return 1.0f / (1.0f + __expf(-x));
}
__device__ __forceinline__ float fast_tanh(float x) {
  x = fminf(fmaxf(x, -15.0f), 15.0f);
  float e = __expf(2.0f * x);
  return (e - 1.0f) / (e + 1.0f);
}

// ---------------------------------------------------------------------------
// Utility kernels
// ---------------------------------------------------------------------------
__global__ void f32_to_bf16(const float* __restrict__ src, bf16_t* __restrict__ dst,
                            long long n) {
  long long i = (long long)blockIdx.x * blockDim.x + threadIdx.x;
  long long stride = (long long)gridDim.x * blockDim.x;
  for (; i < n; i += stride) dst[i] = (bf16_t)src[i];
}

__global__ void zero_u32(unsigned* __restrict__ p, long long n) {
  long long i = (long long)blockIdx.x * blockDim.x + threadIdx.x;
  long long stride = (long long)gridDim.x * blockDim.x;
  for (; i < n; i += stride) p[i] = 0u;
}

// ---------------------------------------------------------------------------
// GEMM: C[M,N] = A[M,K](bf16) * W[N,K](bf16)^T + bias[N], optional ReLU.
// Block = 256 threads = 8 waves. Each wave: 2 M-tiles x 4 N-tiles (8 WMMA accs).
// Block tile = 256 rows x 64 cols. grid = (M/256, N/64).
// The 64-row B panel (shared by all 8 waves) is staged through LDS in KC-column
// chunks; the k-loop then issues only A loads to VMEM + ds reads for B.
// ---------------------------------------------------------------------------
template <int K>
__global__ void gemm_bf16_bias(const bf16_t* __restrict__ A,
                               const bf16_t* __restrict__ W,
                               const float*  __restrict__ bias,
                               float* __restrict__ C,
                               int M, int N, int relu) {
  __shared__ __attribute__((aligned(32))) bf16_t lds_b[64 * KC]; // 32 KB

  const int tid  = threadIdx.x;
  const int wv   = tid >> 5;
  const int lane = tid & 31;
  const int lr   = lane & 15;
  const int g    = lane >> 4;

  const int mt0  = (blockIdx.x * 8 + wv) * 2; // two consecutive M tiles per wave
  const int nrow = blockIdx.y * 64;           // first of 64 B-panel rows (4 N tiles)

  const v8f vzero = {};
  v8f acc[2][4];
#pragma unroll
  for (int i = 0; i < 2; ++i)
#pragma unroll
    for (int j = 0; j < 4; ++j) acc[i][j] = vzero;

  const bf16_t* arow0 = A + (size_t)(mt0 * 16 + lr) * K;
  const bf16_t* arow1 = A + (size_t)((mt0 + 1) * 16 + lr) * K;
  const bf16_t* wpan  = W + (size_t)nrow * K;

#pragma unroll
  for (int k0 = 0; k0 < K; k0 += KC) {
    __syncthreads(); // previous chunk fully consumed
    // Stage 64 x KC bf16 panel: row r, cols [k0, k0+KC) -> lds_b[r*KC ..]
#pragma unroll
    for (int i = tid; i < 64 * KC / 8; i += 256) {
      const int r  = i >> 5;        // / (KC/8)
      const int c8 = i & 31;        // % (KC/8)
      *(v8bf*)&lds_b[r * KC + c8 * 8] =
          *(const v8bf*)(wpan + (size_t)r * K + k0 + c8 * 8);
    }
    __syncthreads();

#pragma unroll
    for (int kt = 0; kt < KC / 32; ++kt) {
      const int ko = k0 + kt * 32;
      v16bf a0 = load_a_frag(arow0 + ko, g);
      v16bf a1 = load_a_frag(arow1 + ko, g);
#pragma unroll
      for (int n = 0; n < 4; ++n) {
        v16bf b = load_b_frag(&lds_b[(n * 16 + lr) * KC + kt * 32], g);
        acc[0][n] = WMMA_BF16(a0, b, acc[0][n]);
        acc[1][n] = WMMA_BF16(a1, b, acc[1][n]);
      }
    }
  }

  float bv[4];
#pragma unroll
  for (int n = 0; n < 4; ++n) bv[n] = bias[nrow + n * 16 + lr];

#pragma unroll
  for (int mi = 0; mi < 2; ++mi) {
#pragma unroll
    for (int n = 0; n < 4; ++n) {
      const int col = nrow + n * 16 + lr;
#pragma unroll
      for (int jv = 0; jv < 8; ++jv) {
        const int row = (mt0 + mi) * 16 + jv + 8 * g;
        float v = acc[mi][n][jv] + bv[n];
        if (relu) v = fmaxf(v, 0.0f);
        C[(size_t)row * N + col] = v;
      }
    }
  }
}

// ---------------------------------------------------------------------------
// Persistent GRU recurrence over T steps.
//   gx  : [B*T, 3H] fp32 (precomputed x@W_ih^T + b_ih), row = b*T + t
//   Whh : [3H, H] fp32 (converted to bf16 into LDS slice per block)
//   hbuf: [2][B, H] bf16 double-buffered broadcast state
//   hseq: [B*T, H] bf16 layer output (feeds next GEMM)
// 32 blocks x 64 threads (2 waves). Block jt owns hidden cols [16*jt, 16*jt+16).
// Wave wv owns batch rows [16*wv, 16*wv+16). h kept fp32 in registers.
// Hot loop: rolling one-k-tile-ahead double buffer of fragments (A from the
// global h broadcast, 3 B's from LDS) so the three WMMAs of tile kt cover the
// latency of tile kt+1's loads -- low register pressure, no spills, no full
// dscnt/loadcnt drains per WMMA.
// ---------------------------------------------------------------------------
__global__ void gru_scan(const float* __restrict__ gx,
                         const float* __restrict__ Whh,
                         const float* __restrict__ bhh,
                         bf16_t* __restrict__ hbuf,
                         bf16_t* __restrict__ hseq,
                         unsigned* __restrict__ counter) {
  __shared__ __attribute__((aligned(32))) bf16_t lds_w[3 * 16 * Hh]; // 48 KB

  const int tid = threadIdx.x;
  const int jt  = blockIdx.x;
  const int j0  = jt * 16;

  // Stage this block's W_hh slice (rows j0..j0+15 of each gate) into LDS as bf16.
  for (int i = tid; i < 3 * 16 * Hh; i += 64) {
    const int gate = i / (16 * Hh);
    const int rem  = i - gate * (16 * Hh);
    const int r    = rem >> 9;        // / 512
    const int c    = rem & (Hh - 1);  // % 512
    lds_w[i] = (bf16_t)Whh[(size_t)(gate * Hh + j0 + r) * Hh + c];
  }
  __syncthreads();

  const int wv   = tid >> 5;
  const int lane = tid & 31;
  const int lr   = lane & 15;
  const int g    = lane >> 4;
  const int m0   = wv * 16;

  const float bhhR = bhh[j0 + lr];
  const float bhhZ = bhh[Hh + j0 + lr];
  const float bhhN = bhh[2 * Hh + j0 + lr];

  const bf16_t* ldsR = &lds_w[(0 * 16 + lr) * Hh];
  const bf16_t* ldsZ = &lds_w[(1 * 16 + lr) * Hh];
  const bf16_t* ldsN = &lds_w[(2 * 16 + lr) * Hh];

  float hr[8];
  const float* gp[8];   // per-acc-element gx row pointer (advances G3 per step)
  bf16_t* sp[8];        // per-acc-element hseq pointer  (advances Hh per step)
#pragma unroll
  for (int jv = 0; jv < 8; ++jv) {
    hr[jv] = 0.0f;
    const int b = m0 + 8 * g + jv;
    gp[jv] = gx + (size_t)b * Tt * G3 + j0 + lr;
    sp[jv] = hseq + (size_t)b * Tt * Hh + j0 + lr;
  }

  const v8f vzero = {};

  for (int t = 0; t < Tt; ++t) {
    const bf16_t* hread  = hbuf + (size_t)(t & 1) * (Bb * Hh);
    bf16_t*       hwrite = hbuf + (size_t)((t + 1) & 1) * (Bb * Hh);

    // Issue gx loads early so they overlap the WMMA chain.
    float gr[8], gz[8], gn[8];
#pragma unroll
    for (int jv = 0; jv < 8; ++jv) {
      gr[jv] = gp[jv][0];
      gz[jv] = gp[jv][Hh];
      gn[jv] = gp[jv][2 * Hh];
    }

    const bf16_t* arow = hread + (size_t)(m0 + lr) * Hh;

    // Rolling double buffer: fragments for tile kt+1 are loaded before the
    // WMMAs of tile kt, hiding LDS/VMEM latency without register blowup.
    v8f aR = vzero, aZ = vzero, aN = vzero;
    v16bf a0 = load_a_frag(arow, g);
    v16bf br = load_b_frag(ldsR, g);
    v16bf bz = load_b_frag(ldsZ, g);
    v16bf bn = load_b_frag(ldsN, g);
#pragma unroll
    for (int kt = 0; kt < Hh / 32; ++kt) {
      v16bf a1, br1, bz1, bn1;
      if (kt + 1 < Hh / 32) {
        const int ko = (kt + 1) * 32;
        a1  = load_a_frag(arow + ko, g);
        br1 = load_b_frag(ldsR + ko, g);
        bz1 = load_b_frag(ldsZ + ko, g);
        bn1 = load_b_frag(ldsN + ko, g);
      }
      aR = WMMA_BF16(a0, br, aR);
      aZ = WMMA_BF16(a0, bz, aZ);
      aN = WMMA_BF16(a0, bn, aN);
      if (kt + 1 < Hh / 32) {
        a0 = a1; br = br1; bz = bz1; bn = bn1;
      }
    }

#pragma unroll
    for (int jv = 0; jv < 8; ++jv) {
      const int b = m0 + 8 * g + jv;
      const float r  = fast_sigmoid(gr[jv] + aR[jv] + bhhR);
      const float z  = fast_sigmoid(gz[jv] + aZ[jv] + bhhZ);
      const float n  = fast_tanh(gn[jv] + r * (aN[jv] + bhhN));
      const float hn = (1.0f - z) * n + z * hr[jv];
      hr[jv] = hn;
      const bf16_t hb = (bf16_t)hn;
      hwrite[(size_t)b * Hh + j0 + lr] = hb;
      *sp[jv] = hb;
      sp[jv] += Hh;
    }

    // Prefetch next step's gx cachelines; global_prefetch does not increment
    // LOADcnt, so the fence below does not wait on it -> HBM latency hides
    // behind the barrier spin (lines land in L2; WGP$ is invalidated anyway).
    if (t + 1 < Tt) {
#pragma unroll
      for (int jv = 0; jv < 8; ++jv) {
        __builtin_prefetch(gp[jv] + G3, 0, 1);
        __builtin_prefetch(gp[jv] + G3 + Hh, 0, 1);
        __builtin_prefetch(gp[jv] + G3 + 2 * Hh, 0, 1);
      }
    }
#pragma unroll
    for (int jv = 0; jv < 8; ++jv) gp[jv] += G3;

    // Grid barrier: all 32 blocks must publish h(t+1) before anyone reads it.
    __syncthreads();
    __threadfence();
    if (tid == 0) {
      atomicAdd(counter, 1u);
      const unsigned target = (unsigned)NB_SCAN * (unsigned)(t + 1);
      while (__hip_atomic_load(counter, __ATOMIC_RELAXED, __HIP_MEMORY_SCOPE_AGENT) < target) {
        __builtin_amdgcn_s_sleep(1);
      }
    }
    __syncthreads();
    __threadfence();
  }
}

// ---------------------------------------------------------------------------
// Host-side launch
// ---------------------------------------------------------------------------
extern "C" void kernel_launch(void* const* d_in, const int* in_sizes, int n_in,
                              void* d_out, int out_size, void* d_ws, size_t ws_size,
                              hipStream_t stream) {
  (void)in_sizes; (void)n_in; (void)out_size; (void)ws_size;

  const float* x    = (const float*)d_in[0];
  const float* Wih0 = (const float*)d_in[1];
  const float* Whh0 = (const float*)d_in[2];
  const float* bih0 = (const float*)d_in[3];
  const float* bhh0 = (const float*)d_in[4];
  const float* Wih1 = (const float*)d_in[5];
  const float* Whh1 = (const float*)d_in[6];
  const float* bih1 = (const float*)d_in[7];
  const float* bhh1 = (const float*)d_in[8];
  const float* Wf   = (const float*)d_in[9];
  const float* bfv  = (const float*)d_in[10];
  float* out = (float*)d_out;

  char* ws = (char*)d_ws;
  size_t off = 0;
  auto walloc = [&](size_t bytes) -> char* {
    char* p = ws + off;
    off += (bytes + 255) & ~(size_t)255;
    return p;
  };

  float*  gx    = (float*)walloc((size_t)Bb * Tt * G3 * 4);  // 201 MB, reused per layer
  bf16_t* xb    = (bf16_t*)walloc((size_t)Bb * Tt * Dd * 2);
  bf16_t* h1b   = (bf16_t*)walloc((size_t)Bb * Tt * Hh * 2);
  bf16_t* h2b   = (bf16_t*)walloc((size_t)Bb * Tt * Hh * 2);
  bf16_t* wih0b = (bf16_t*)walloc((size_t)G3 * Dd * 2);
  bf16_t* wih1b = (bf16_t*)walloc((size_t)G3 * Hh * 2);
  bf16_t* wfb   = (bf16_t*)walloc((size_t)Oo * Hh * 2);
  bf16_t* hbuf  = (bf16_t*)walloc((size_t)2 * Bb * Hh * 2);
  unsigned* ctr = (unsigned*)walloc(256);

  // Convert inputs / weights to bf16.
  f32_to_bf16<<<256, 256, 0, stream>>>(x,    xb,    (long long)Bb * Tt * Dd);
  f32_to_bf16<<<128, 256, 0, stream>>>(Wih0, wih0b, (long long)G3 * Dd);
  f32_to_bf16<<<128, 256, 0, stream>>>(Wih1, wih1b, (long long)G3 * Hh);
  f32_to_bf16<<<64,  256, 0, stream>>>(Wf,   wfb,   (long long)Oo * Hh);

  // Zero h state (both buffers) + barrier counters: deterministic per launch.
  zero_u32<<<64, 256, 0, stream>>>((unsigned*)hbuf, (long long)(2 * Bb * Hh * 2 / 4));
  zero_u32<<<1, 64, 0, stream>>>(ctr, 64);

  const dim3 blk(256);

  // Layer 0: gx0 = xb @ Wih0^T + bih0   [32768 x 1536], K=256
  gemm_bf16_bias<Dd><<<dim3(Bb * Tt / 256, G3 / 64), blk, 0, stream>>>(
      xb, wih0b, bih0, gx, Bb * Tt, G3, 0);
  gru_scan<<<NB_SCAN, 64, 0, stream>>>(gx, Whh0, bhh0, hbuf, h1b, ctr + 0);

  // Reset h state for layer 1.
  zero_u32<<<64, 256, 0, stream>>>((unsigned*)hbuf, (long long)(2 * Bb * Hh * 2 / 4));

  // Layer 1: gx1 = h1 @ Wih1^T + bih1   [32768 x 1536], K=512
  gemm_bf16_bias<Hh><<<dim3(Bb * Tt / 256, G3 / 64), blk, 0, stream>>>(
      h1b, wih1b, bih1, gx, Bb * Tt, G3, 0);
  gru_scan<<<NB_SCAN, 64, 0, stream>>>(gx, Whh1, bhh1, hbuf, h2b, ctr + 1);

  // Final: out = relu(h2 @ Wf^T + bf)   [32768 x 256], K=512
  gemm_bf16_bias<Hh><<<dim3(Bb * Tt / 256, Oo / 64), blk, 0, stream>>>(
      h2b, wfb, bfv, out, Bb * Tt, Oo, 1);
}
